// TimeMix_15298673508626
// MI455X (gfx1250) — compile-verified, matrix operand order checked by reading, not run on hardware
//
#include <hip/hip_runtime.h>

// ---------------- problem constants ----------------
#define B_    4
#define T_    1024
#define D_    1024
#define H_    16
#define HS_   64
#define DM_   1024
#define N_TOK (B_ * T_)             // 4096 tokens
#define ND    ((size_t)N_TOK * D_)  // 4 Mi elements
#define EPS_      1e-5f
#define DECAY_C_  (-0.606531f)

typedef __attribute__((ext_vector_type(16))) __bf16 bf16x16;
typedef __attribute__((ext_vector_type(8)))  float  f32x8;

// ---------------- helpers ----------------
__device__ __forceinline__ unsigned short f2bf(float f) {
  unsigned int u = __builtin_bit_cast(unsigned int, f);
  u += 0x7FFFu + ((u >> 16) & 1u);          // round-to-nearest-even
  return (unsigned short)(u >> 16);
}

__device__ __forceinline__ float sigmoidf_(float x) {
  return 1.0f / (1.0f + __expf(-x));
}

// Low 32 bits of a generic shared-memory pointer == LDS byte offset
// (ISA 10.2: flat addresses in the LDS aperture use their low 32 bits as the LDS address).
__device__ __forceinline__ unsigned lds_off(const void* p) {
  return (unsigned)(size_t)p;
}

// ---------------- fp32 -> bf16 conversion ----------------
__global__ void cvt_f32_bf16(const float* __restrict__ src,
                             unsigned short* __restrict__ dst, int n) {
  int i = blockIdx.x * blockDim.x + threadIdx.x;
  if (i < n) dst[i] = f2bf(src[i]);
}

// ---------------- token-shift mixing: x6 = cur + rwkvag[j]*(prev-cur) ----------------
__global__ void mix_kernel(const float* __restrict__ cur,
                           const float* __restrict__ prev,
                           const float* __restrict__ rwkvag,
                           unsigned short* __restrict__ xall) {
  size_t i = (size_t)blockIdx.x * blockDim.x + threadIdx.x;
  if (i >= ND) return;
  int d = (int)(i & (D_ - 1));
  float c  = cur[i];
  float df = prev[i] - c;
#pragma unroll
  for (int j = 0; j < 6; ++j)
    xall[(size_t)j * ND + i] = f2bf(c + rwkvag[j * D_ + d] * df);
}

// ---------------- WMMA bf16 GEMM: C[N x M] = A[N x K] @ W[K x M] (+bias)(+act) ------
// Block tile 128x128, 8 waves (4 row-groups x 2 col-groups), each wave 32x64 = 2x4
// 16x16 frags -> 8 v_wmma_f32_16x16x32_bf16 per K-step (K-step 32).
// Double-buffered LDS: tile i+1 staged (A via global_load_async_to_lds_b128 / ASYNCcnt,
// B transposed with packed ds_store_b32 pairs) while tile i is consumed.
// act: 0=none 1=tanh 2=sigmoid 3=exp(DECAY_C * sigmoid(x))
__global__ void __launch_bounds__(256)
gemm_bf16(const unsigned short* __restrict__ A,
          const unsigned short* __restrict__ W,
          const float* __restrict__ bias,
          float* __restrict__ outF,
          unsigned short* __restrict__ outB,
          int K, int M, int act) {
  __shared__ unsigned short As[2][128 * 32];    // 2 x 8 KB, [row][k]
  __shared__ unsigned short Bst[2][128 * 32];   // 2 x 8 KB, [col][k] (transposed)

  const int tid  = threadIdx.x;
  const int lane = tid & 31;
  const int wid  = tid >> 5;
  const int wm   = (wid & 3) << 5;              // wave row offset (4 x 32)
  const int wn   = (wid >> 2) << 6;             // wave col offset (2 x 64)
  const int m0   = blockIdx.y * 128;
  const int n0   = blockIdx.x * 128;

  const int mrow = lane & 15;                   // row (A) / col (B) within fragment
  const int kb   = (lane >> 4) << 3;            // lane-group K offset (0 or 8)

  // ---- constant per-thread staging coordinates ----
  // A: two 8-half chunks per thread
  const int arow0 = tid >> 2;                   // 0..63
  const int arow1 = arow0 + 64;                 // 64..127
  const int acol  = (tid & 3) << 3;             // 0,8,16,24
  const unsigned short* agp0 = A + (size_t)(m0 + arow0) * K + acol;
  const unsigned short* agp1 = A + (size_t)(m0 + arow1) * K + acol;
  // B: one chunk of 2 k-rows x 8 cols per thread
  const int kr2  = (tid >> 4) << 1;             // even k-row 0..30
  const int bcol = (tid & 15) << 3;             // 0..120
  const bool bvalid = (n0 + bcol + 8) <= M;     // M is always a multiple of 8
  const unsigned short* bgp = W + (size_t)kr2 * M + (n0 + bcol);

  auto stage = [&](int kt, int buf) {
    // A tile 128x32 via async copy to LDS (2 x b128 per thread, ASYNCcnt)
    {
      const unsigned short* g0 = agp0 + kt;
      const unsigned short* g1 = agp1 + kt;
      unsigned lo0 = lds_off(&As[buf][arow0 * 32 + acol]);
      unsigned lo1 = lds_off(&As[buf][arow1 * 32 + acol]);
      asm volatile("global_load_async_to_lds_b128 %0, %1, off"
                   :: "v"(lo0), "v"(g0) : "memory");
      asm volatile("global_load_async_to_lds_b128 %0, %1, off"
                   :: "v"(lo1), "v"(g1) : "memory");
    }
    // B tile 32x128 -> transposed LDS; pack k-row pairs -> ds_store_b32
    if (bvalid) {
      const unsigned short* g = bgp + (size_t)kt * M;
      uint4 d0 = *(const uint4*)g;
      uint4 d1 = *(const uint4*)(g + M);
      unsigned short lo8[8], hi8[8];
      __builtin_memcpy(lo8, &d0, 16);
      __builtin_memcpy(hi8, &d1, 16);
#pragma unroll
      for (int j = 0; j < 8; ++j) {
        unsigned pk = (unsigned)lo8[j] | ((unsigned)hi8[j] << 16);
        *(unsigned*)&Bst[buf][(bcol + j) * 32 + kr2] = pk;
      }
    }
  };

  f32x8 acc[2][4] = {};

  stage(0, 0);
  asm volatile("s_wait_asynccnt 0x0" ::: "memory");
  __syncthreads();

  int buf = 0;
  for (int kt = 0; kt < K; kt += 32) {
    const bool more = (kt + 32) < K;
    if (more) {
      stage(kt + 32, buf ^ 1);                   // overlap with compute below
      __builtin_prefetch(agp0 + kt + 64, 0, 1);  // speculative, next-next tile
      __builtin_prefetch(bgp + (size_t)(kt + 64) * M, 0, 1);
    }

    // ---- fragments: contiguous 8-half groups -> ds_load_b128 ----
    bf16x16 afr[2], bfr[4];
#pragma unroll
    for (int fm = 0; fm < 2; ++fm) {
      unsigned short tmp[16];
      int row = wm + fm * 16 + mrow;
#pragma unroll
      for (int h = 0; h < 16; ++h) {
        int k = ((h >> 3) << 4) + kb + (h & 7);
        tmp[h] = As[buf][row * 32 + k];
      }
      __builtin_memcpy(&afr[fm], tmp, sizeof(afr[fm]));
    }
#pragma unroll
    for (int fn = 0; fn < 4; ++fn) {
      unsigned short tmp[16];
      int col = wn + fn * 16 + mrow;
#pragma unroll
      for (int h = 0; h < 16; ++h) {
        int k = ((h >> 3) << 4) + kb + (h & 7);
        tmp[h] = Bst[buf][col * 32 + k];
      }
      __builtin_memcpy(&bfr[fn], tmp, sizeof(bfr[fn]));
    }

#pragma unroll
    for (int fm = 0; fm < 2; ++fm)
#pragma unroll
      for (int fn = 0; fn < 4; ++fn)
        acc[fm][fn] = __builtin_amdgcn_wmma_f32_16x16x32_bf16(
            false, afr[fm], false, bfr[fn], (short)0, acc[fm][fn], false, false);

    if (more) {
      asm volatile("s_wait_asynccnt 0x0" ::: "memory");
      __syncthreads();
      buf ^= 1;
    }
  }

  // ---- store (C/D layout: VGPR i -> row i + 8*(lane>=16), col = lane&15) ----
#pragma unroll
  for (int fm = 0; fm < 2; ++fm)
#pragma unroll
    for (int fn = 0; fn < 4; ++fn)
#pragma unroll
      for (int i = 0; i < 8; ++i) {
        int row = m0 + wm + fm * 16 + i + ((lane >> 4) << 3);
        int col = n0 + wn + fn * 16 + (lane & 15);
        if (col < M) {
          float v = acc[fm][fn][i];
          if (bias) v += bias[col];
          if (act == 1)      v = tanhf(v);
          else if (act == 2) v = sigmoidf_(v);
          else if (act == 3) v = __expf(DECAY_C_ * sigmoidf_(v));
          size_t o = (size_t)row * M + col;
          if (outF) outF[o] = v;
          if (outB) outB[o] = f2bf(v);
        }
      }
}

// ---------------- post-GEMM elementwise + per-head reductions ----------------
// In-place reuse:  k0 -> k,  v0 -> v,  a -> bt,  vsig -> at.  Also writes rk_sum.
__global__ void __launch_bounds__(256)
post_kernel(const float* __restrict__ r,
            float* __restrict__ k0, float* __restrict__ v0,
            float* __restrict__ a,  float* __restrict__ vsig,
            const float* __restrict__ v_first,
            const float* __restrict__ k_k, const float* __restrict__ k_a,
            const float* __restrict__ r_k, float* __restrict__ rk_sum) {
  __shared__ float part[256];
  __shared__ float hval[16];
  const int tok = blockIdx.x;
  const int t   = threadIdx.x;
  const size_t base = (size_t)tok * DM_ + t * 4;

  float kkraw[4], aa[4], vs[4];
  float s = 0.f;
#pragma unroll
  for (int j = 0; j < 4; ++j) {
    int c = t * 4 + j;
    kkraw[j] = k0[base + j] * k_k[c];
    s += kkraw[j] * kkraw[j];
    aa[j] = a[base + j];
    vs[j] = vsig[base + j];
  }
  part[t] = s;
  __syncthreads();
  if ((t & 15) == 0) {
    float acc = 0.f;
    for (int q = 0; q < 16; ++q) acc += part[t + q];
    hval[t >> 4] = rsqrtf(acc + 1e-12f);     // 1/||kk|| per head
  }
  __syncthreads();
  const int head = t >> 4;
  const float inv = hval[head];

  float s2 = 0.f;
#pragma unroll
  for (int j = 0; j < 4; ++j) {
    int c = t * 4 + j;
    float kk   = kkraw[j] * inv;
    float knew = k0[base + j] * (1.f + (aa[j] - 1.f) * k_a[c]);
    float vnew = v0[base + j] + (v_first[base + j] - v0[base + j]) * vs[j];
    k0[base + j]   = knew;          // k
    v0[base + j]   = vnew;          // v
    vsig[base + j] = -kk;           // at
    a[base + j]    = kk * aa[j];    // bt
    s2 += r[base + j] * knew * r_k[c];
  }
  part[t] = s2;
  __syncthreads();
  if ((t & 15) == 0) {
    float acc = 0.f;
    for (int q = 0; q < 16; ++q) acc += part[t + q];
    rk_sum[tok * H_ + (t >> 4)] = acc;
  }
}

// ---------------- sequential T-scan: one block per (b,h), state rows in VGPRs ----
// Double-buffered LDS: next step's vectors loaded from global while current step
// computes, one barrier per step.
__global__ void __launch_bounds__(64)
scan_kernel(const float* __restrict__ r, const float* __restrict__ w,
            const float* __restrict__ k, const float* __restrict__ v,
            const float* __restrict__ at, const float* __restrict__ bt,
            float* __restrict__ y, float* __restrict__ state_out) {
  const int bh = blockIdx.x;                 // 0 .. B*H-1
  const int b = bh >> 4, h = bh & 15;
  const int tid = threadIdx.x;               // v-row index 0..63
  __shared__ float sh[2][6][64];             // [buf][r,w,k,v,at,bt][chan]

  float st[HS_];
#pragma unroll
  for (int i = 0; i < HS_; ++i) st[i] = 0.f;

  const size_t base = ((size_t)b * T_) * DM_ + h * HS_ + tid;

  // preload t = 0
  {
    const size_t o = base;
    sh[0][0][tid] = r[o];  sh[0][1][tid] = w[o];  sh[0][2][tid] = k[o];
    sh[0][3][tid] = v[o];  sh[0][4][tid] = at[o]; sh[0][5][tid] = bt[o];
  }
  __syncthreads();

  int cur = 0;
  for (int t = 0; t < T_; ++t) {
    float lr = 0.f, lw = 0.f, lk = 0.f, lv = 0.f, la = 0.f, lb = 0.f;
    const bool more = (t + 1) < T_;
    if (more) {                              // issue next-step loads early
      const size_t o2 = base + (size_t)(t + 1) * DM_;
      lr = r[o2];  lw = w[o2];  lk = k[o2];
      lv = v[o2];  la = at[o2]; lb = bt[o2];
    }

    float sav = 0.f;
#pragma unroll
    for (int q = 0; q < HS_; ++q) sav += st[q] * sh[cur][4][q];
    const float vown = sh[cur][3][tid];
    float yv = 0.f;
#pragma unroll
    for (int q = 0; q < HS_; ++q) {
      st[q] = st[q] * sh[cur][1][q] + sav * sh[cur][5][q] + vown * sh[cur][2][q];
      yv += st[q] * sh[cur][0][q];
    }
    y[base + (size_t)t * DM_] = yv;

    if (more) {
      sh[cur ^ 1][0][tid] = lr;  sh[cur ^ 1][1][tid] = lw;  sh[cur ^ 1][2][tid] = lk;
      sh[cur ^ 1][3][tid] = lv;  sh[cur ^ 1][4][tid] = la;  sh[cur ^ 1][5][tid] = lb;
      __syncthreads();
      cur ^= 1;
    }
  }

  // final state: (B,H,HS,HS) with state[b,h,v,k]
  const size_t so = ((size_t)(b * H_ + h) * HS_ + tid) * HS_;
#pragma unroll
  for (int q = 0; q < HS_; ++q) state_out[so + q] = st[q];
}

// ---------------- GroupNorm (per head) + rk_sum*v + gate -> bf16 ----------------
__global__ void __launch_bounds__(256)
gn_kernel(const float* __restrict__ y, const float* __restrict__ v,
          const float* __restrict__ g, const float* __restrict__ rk_sum,
          const float* __restrict__ gn_scale, const float* __restrict__ gn_bias,
          unsigned short* __restrict__ z) {
  __shared__ float p1[256], p2[256];
  __shared__ float mus[16], rvs[16];
  const int tok = blockIdx.x;
  const int t   = threadIdx.x;
  const size_t base = (size_t)tok * DM_ + t * 4;

  float yv[4];
  float s1 = 0.f, s2 = 0.f;
#pragma unroll
  for (int j = 0; j < 4; ++j) {
    yv[j] = y[base + j];
    s1 += yv[j];
    s2 += yv[j] * yv[j];
  }
  p1[t] = s1; p2[t] = s2;
  __syncthreads();
  if ((t & 15) == 0) {
    float a1s = 0.f, a2s = 0.f;
    for (int q = 0; q < 16; ++q) { a1s += p1[t + q]; a2s += p2[t + q]; }
    float mu  = a1s * (1.f / HS_);
    float var = a2s * (1.f / HS_) - mu * mu;
    mus[t >> 4] = mu;
    rvs[t >> 4] = rsqrtf(var + EPS_);
  }
  __syncthreads();
  const int head = t >> 4;
  const float mu = mus[head], rv = rvs[head];
  const float rks = rk_sum[tok * H_ + head];
#pragma unroll
  for (int j = 0; j < 4; ++j) {
    int c = t * 4 + j;
    float o = (yv[j] - mu) * rv * gn_scale[c] + gn_bias[c];
    o += rks * v[base + j];
    z[base + j] = f2bf(o * g[base + j]);
  }
}

// ---------------- plain f32 copy (v_first passthrough output) ----------------
__global__ void copy_kernel(const float* __restrict__ src,
                            float* __restrict__ dst, int n) {
  int i = blockIdx.x * blockDim.x + threadIdx.x;
  if (i < n) dst[i] = src[i];
}

// =======================================================================
extern "C" void kernel_launch(void* const* d_in, const int* in_sizes, int n_in,
                              void* d_out, int out_size, void* d_ws, size_t ws_size,
                              hipStream_t stream) {
  // ---- inputs (setup_inputs order) ----
  const float* current  = (const float*)d_in[0];
  const float* prev     = (const float*)d_in[1];
  const float* v_first  = (const float*)d_in[2];
  const float* rwkvag   = (const float*)d_in[3];
  const float* rw_W     = (const float*)d_in[4];
  const float* w1       = (const float*)d_in[5];
  const float* w2       = (const float*)d_in[6];
  const float* w2_bias  = (const float*)d_in[7];
  const float* kw_W     = (const float*)d_in[8];
  const float* vw_W     = (const float*)d_in[9];
  const float* a1       = (const float*)d_in[10];
  const float* a2       = (const float*)d_in[11];
  const float* a2_bias  = (const float*)d_in[12];
  const float* v1       = (const float*)d_in[13];
  const float* v2       = (const float*)d_in[14];
  const float* v2_bias  = (const float*)d_in[15];
  const float* g1       = (const float*)d_in[16];
  const float* g2       = (const float*)d_in[17];
  const float* g2_bias  = (const float*)d_in[18];
  const float* k_k      = (const float*)d_in[19];
  const float* k_a      = (const float*)d_in[20];
  const float* r_k      = (const float*)d_in[21];
  const float* gn_scale = (const float*)d_in[22];
  const float* gn_bias  = (const float*)d_in[23];
  const float* out_W    = (const float*)d_in[24];

  float* out_y     = (float*)d_out;                               // (B,T,D)
  float* out_state = (float*)d_out + (size_t)N_TOK * D_;          // (B,H,HS,HS)
  float* out_vf    = out_state + (size_t)B_ * H_ * HS_ * HS_;     // (B,T,DM)

  // ---- workspace bump allocator ----
  char* p = (char*)d_ws;
  auto alloc = [&](size_t bytes) -> void* {
    void* r = (void*)p;
    p += (bytes + 255) & ~(size_t)255;
    return r;
  };
  unsigned short* xall  = (unsigned short*)alloc(6 * ND * 2);     // xr,xw,xk,xv,xa,xg bf16
  unsigned short* rw_b  = (unsigned short*)alloc((size_t)D_ * D_ * 2);
  unsigned short* kw_b  = (unsigned short*)alloc((size_t)D_ * DM_ * 2);
  unsigned short* vw_b  = (unsigned short*)alloc((size_t)D_ * DM_ * 2);
  unsigned short* ow_b  = (unsigned short*)alloc((size_t)DM_ * D_ * 2);
  unsigned short* w1_b  = (unsigned short*)alloc((size_t)D_ * 64 * 2);
  unsigned short* w2_b  = (unsigned short*)alloc((size_t)64 * DM_ * 2);
  unsigned short* a1_b  = (unsigned short*)alloc((size_t)D_ * 64 * 2);
  unsigned short* a2_b  = (unsigned short*)alloc((size_t)64 * DM_ * 2);
  unsigned short* v1_b  = (unsigned short*)alloc((size_t)D_ * 32 * 2);
  unsigned short* v2_b  = (unsigned short*)alloc((size_t)32 * DM_ * 2);
  unsigned short* g1_b  = (unsigned short*)alloc((size_t)D_ * 128 * 2);
  unsigned short* g2_b  = (unsigned short*)alloc((size_t)128 * DM_ * 2);
  unsigned short* h_w   = (unsigned short*)alloc((size_t)N_TOK * 64 * 2);
  unsigned short* h_a   = (unsigned short*)alloc((size_t)N_TOK * 64 * 2);
  unsigned short* h_v   = (unsigned short*)alloc((size_t)N_TOK * 32 * 2);
  unsigned short* h_g   = (unsigned short*)alloc((size_t)N_TOK * 128 * 2);
  float* rbuf   = (float*)alloc(ND * 4);
  float* kbuf   = (float*)alloc(ND * 4);   // k0 -> k
  float* vbuf   = (float*)alloc(ND * 4);   // v0 -> v
  float* wdec   = (float*)alloc(ND * 4);
  float* abuf   = (float*)alloc(ND * 4);   // a  -> bt
  float* atbuf  = (float*)alloc(ND * 4);   // vsig -> at
  float* gbuf   = (float*)alloc(ND * 4);
  float* ybuf   = (float*)alloc(ND * 4);
  float* rksum  = (float*)alloc((size_t)N_TOK * H_ * 4);
  unsigned short* zbuf = (unsigned short*)alloc(ND * 2);

  // ---- 1) weight conversion fp32 -> bf16 ----
  auto cvt = [&](const float* s, unsigned short* d, int n) {
    cvt_f32_bf16<<<dim3((n + 255) / 256), dim3(256), 0, stream>>>(s, d, n);
  };
  cvt(rw_W, rw_b, D_ * D_);
  cvt(kw_W, kw_b, D_ * DM_);
  cvt(vw_W, vw_b, D_ * DM_);
  cvt(out_W, ow_b, DM_ * D_);
  cvt(w1, w1_b, D_ * 64);   cvt(w2, w2_b, 64 * DM_);
  cvt(a1, a1_b, D_ * 64);   cvt(a2, a2_b, 64 * DM_);
  cvt(v1, v1_b, D_ * 32);   cvt(v2, v2_b, 32 * DM_);
  cvt(g1, g1_b, D_ * 128);  cvt(g2, g2_b, 128 * DM_);

  // ---- 2) token-shift mix ----
  mix_kernel<<<dim3((unsigned)((ND + 255) / 256)), dim3(256), 0, stream>>>(
      current, prev, rwkvag, xall);

  const unsigned short* xr = xall + 0 * ND;
  const unsigned short* xw = xall + 1 * ND;
  const unsigned short* xk = xall + 2 * ND;
  const unsigned short* xv = xall + 3 * ND;
  const unsigned short* xa = xall + 4 * ND;
  const unsigned short* xg = xall + 5 * ND;

  auto gemm = [&](const unsigned short* A, const unsigned short* W,
                  const float* bias, float* oF, unsigned short* oB,
                  int K, int M, int act) {
    dim3 grid((M + 127) / 128, N_TOK / 128);
    gemm_bf16<<<grid, dim3(256), 0, stream>>>(A, W, bias, oF, oB, K, M, act);
  };

  // ---- 3) stage-1 GEMMs ----
  gemm(xr, rw_b, nullptr, rbuf, nullptr, D_, D_, 0);        // r
  gemm(xk, kw_b, nullptr, kbuf, nullptr, D_, DM_, 0);       // k0
  gemm(xv, vw_b, nullptr, vbuf, nullptr, D_, DM_, 0);       // v0
  gemm(xw, w1_b, nullptr, nullptr, h_w, D_, 64, 1);         // tanh(xw@w1)
  gemm(xa, a1_b, nullptr, nullptr, h_a, D_, 64, 0);         // xa@a1
  gemm(xv, v1_b, nullptr, nullptr, h_v, D_, 32, 0);         // xv@v1
  gemm(xg, g1_b, nullptr, nullptr, h_g, D_, 128, 2);        // sigmoid(xg@g1)

  // ---- 4) stage-2 LoRA GEMMs ----
  gemm(h_w, w2_b, w2_bias, wdec, nullptr, 64, DM_, 3);      // exp(C*sigmoid(wl))
  gemm(h_a, a2_b, a2_bias, abuf, nullptr, 64, DM_, 2);      // a = sigmoid(.)
  gemm(h_v, v2_b, v2_bias, atbuf, nullptr, 32, DM_, 2);     // vsig (becomes at)
  gemm(h_g, g2_b, g2_bias, gbuf, nullptr, 128, DM_, 0);     // g

  // ---- 5) elementwise post + per-head reductions ----
  post_kernel<<<dim3(N_TOK), dim3(256), 0, stream>>>(
      rbuf, kbuf, vbuf, abuf, atbuf, v_first, k_k, k_a, r_k, rksum);

  // ---- 6) sequential scan over T (also writes final state output) ----
  scan_kernel<<<dim3(B_ * H_), dim3(HS_), 0, stream>>>(
      rbuf, wdec, kbuf, vbuf, atbuf, abuf, ybuf, out_state);

  // ---- 7) GroupNorm + rk_sum*v + gate ----
  gn_kernel<<<dim3(N_TOK), dim3(256), 0, stream>>>(
      ybuf, vbuf, gbuf, rksum, gn_scale, gn_bias, zbuf);

  // ---- 8) output projection ----
  gemm(zbuf, ow_b, nullptr, out_y, nullptr, DM_, D_, 0);

  // ---- 9) v_first passthrough output ----
  copy_kernel<<<dim3((unsigned)((ND + 255) / 256)), dim3(256), 0, stream>>>(
      v_first, out_vf, (int)ND);

  (void)in_sizes; (void)n_in; (void)out_size; (void)ws_size;
}
